// MultiHeadAttention_76089640616629
// MI455X (gfx1250) — compile-verified, matrix operand order checked by reading, not run on hardware
//
#include <hip/hip_runtime.h>
#include <hip/hip_bf16.h>

// MHA: B=4, S=2048, D=768, H=12, DH=64.  All GEMMs on v_wmma_f32_16x16x32_f16.
//  phase 0: weights -> f16 B-fragments (tiny)
//  phase 1: QKV projection via WMMA, outputs pre-swizzled f16 fragments
//  phase 2: flash attention, 16 queries/wave, 128 keys/iteration (32 WMMA/iter)

#define Bq   4
#define Sq   2048
#define Dq   768
#define Hq   12
#define DHq  64
#define NT16  (Sq / 16)    // 128
#define NT32  (Sq / 32)    // 64
#define NT128 (Sq / 128)   // 16
#define QKV_ELEMS ((size_t)Bq * Hq * Sq * DHq)       // 6291456 halfs per tensor
#define WFRAG_ELEMS ((size_t)3 * Hq * 8 * 32 * 16)   // 147456 halfs

typedef __attribute__((ext_vector_type(16))) _Float16 v16h;
typedef __attribute__((ext_vector_type(8)))  _Float16 v8h;
typedef __attribute__((ext_vector_type(8)))  float    v8f;
using h2 = decltype(__builtin_amdgcn_cvt_pkrtz(0.f, 0.f));

// Q/K buffer (A/B fragment layout): [b][h][t16][frag(2)][lane(32)][elem(16)]
__device__ __host__ inline size_t qidx(int b, int h, int t16, int frag, int lane, int elem) {
  return ((((size_t)(b * Hq + h) * NT16 + t16) * 2 + frag) * 32 + lane) * 16 + elem;
}
// V buffer (B fragment layout): [b][h][t32][dhb(4)][lane(32)][elem(16)]
__device__ __host__ inline size_t vidx(int b, int h, int t32, int dhb, int lane, int elem) {
  return ((((size_t)(b * Hq + h) * NT32 + t32) * 4 + dhb) * 32 + lane) * 16 + elem;
}

__device__ inline v8f wmma_f16(v16h a, v16h b, v8f c) {
  return __builtin_amdgcn_wmma_f32_16x16x32_f16(false, a, false, b,
                                                (short)0, c, false, false);
}
__device__ inline v16h cat8(v8h lo, v8h hi) {
  return __builtin_shufflevector(lo, hi, 0, 1, 2, 3, 4, 5, 6, 7,
                                 8, 9, 10, 11, 12, 13, 14, 15);
}
__device__ inline h2 h2max(h2 a, h2 b) {
#if __has_builtin(__builtin_elementwise_max)
  return __builtin_elementwise_max(a, b);   // v_pk_max_num_f16
#else
  h2 r;
  r[0] = a[0] > b[0] ? a[0] : b[0];
  r[1] = a[1] > b[1] ? a[1] : b[1];
  return r;
#endif
}

// ---------------------------------------------------------------------------
// Phase 0: convert W^T (per head, per matrix) to f16 B-fragments.
// gid = ((mat*H + h)*2 + kf)*4 + nblk;  value(lane,e) = w[n][d],
//   n = nblk*16 + lane%16,  d = kf*32 + (e/8)*16 + (lane/16)*8 + e%8.
// ---------------------------------------------------------------------------
__global__ __launch_bounds__(256) void wprep_kernel(
    const float* __restrict__ wq, const float* __restrict__ wk,
    const float* __restrict__ wv, _Float16* __restrict__ wfrag)
{
  const int gid  = blockIdx.x * 8 + (threadIdx.x >> 5);   // 0..287
  const int lane = threadIdx.x & 31;
  const int nblk = gid & 3;
  const int kf   = (gid >> 2) & 1;
  const int hm   = gid >> 3;
  const int h    = hm % Hq;
  const int mat  = hm / Hq;
  const float* w = (mat == 0) ? wq : (mat == 1) ? wk : wv;

  const int n     = nblk * 16 + (lane & 15);
  const int dbase = kf * 32 + (lane >> 4) * 8;
  const float* wrow = w + ((size_t)h * DHq + n) * DHq;

  v16h frag;
#pragma unroll
  for (int o = 0; o < 8; ++o) {
    frag[o]     = (_Float16)wrow[dbase + o];
    frag[8 + o] = (_Float16)wrow[dbase + 16 + o];
  }
  *(v16h*)(wfrag + ((size_t)gid * 32 + lane) * 16) = frag;
}

// ---------------------------------------------------------------------------
// Phase 1: QKV projection via WMMA.  One wave per (b,h,16-row tile).
// ---------------------------------------------------------------------------
template <int MAT>
__device__ inline void project_store(
    int lane, int b, int h, int t16, v16h ax0, v16h ax1,
    const _Float16* __restrict__ wfrag, const float* __restrict__ bias,
    _Float16* __restrict__ dst, _Float16* cht /* wave's 16x72 LDS tile */)
{
  const int l16 = lane & 15, grp = lane >> 4;
  const _Float16* wf = wfrag + (size_t)((MAT * Hq + h) * 8) * 512;

#pragma unroll
  for (int nblk = 0; nblk < 4; ++nblk) {
    v16h b0 = *(const v16h*)(wf + ((size_t)(nblk) * 32 + lane) * 16);
    v16h b1 = *(const v16h*)(wf + ((size_t)(4 + nblk) * 32 + lane) * 16);
    v8f c = {};
    c = wmma_f16(ax0, b0, c);
    c = wmma_f16(ax1, b1, c);
    float bb = bias[h * DHq + nblk * 16 + l16];
#pragma unroll
    for (int i = 0; i < 8; ++i) {
      float v = c[i] + bb;
      if (MAT == 0) v *= 0.125f;              // fold 1/sqrt(64) into Q
      cht[(i + grp * 8) * 72 + nblk * 16 + l16] = (_Float16)v;
    }
  }
  asm volatile("s_wait_dscnt 0" ::: "memory");

  if (MAT < 2) {
    const _Float16* pr = cht + l16 * 72;
    const int hi8 = grp * 8;
    v8h a0 = *(const v8h*)(pr + hi8);
    v8h a1 = *(const v8h*)(pr + 16 + hi8);
    v8h a2 = *(const v8h*)(pr + 32 + hi8);
    v8h a3 = *(const v8h*)(pr + 48 + hi8);
    _Float16* d = dst + qidx(b, h, t16, 0, 0, 0);
    *(v16h*)(d + lane * 16)        = cat8(a0, a1);
    *(v16h*)(d + (32 + lane) * 16) = cat8(a2, a3);
  } else {
    const int t32 = t16 >> 1, halfsel = t16 & 1;
    _Float16* d = dst + vidx(b, h, t32, 0, 0, 0);
#pragma unroll
    for (int dhb = 0; dhb < 4; ++dhb) {
      v8h vv;
#pragma unroll
      for (int o = 0; o < 8; ++o)
        vv[o] = cht[(grp * 8 + o) * 72 + dhb * 16 + l16];
      *(v8h*)(d + (size_t)(dhb * 32 + lane) * 16 + halfsel * 8) = vv;
    }
  }
  asm volatile("s_wait_dscnt 0" ::: "memory");
}

__global__ __launch_bounds__(256) void qkv_kernel(
    const float* __restrict__ seq, const _Float16* __restrict__ wfrag,
    const float* __restrict__ bq, const float* __restrict__ bk,
    const float* __restrict__ bv,
    _Float16* __restrict__ qbuf, _Float16* __restrict__ kbuf,
    _Float16* __restrict__ vbuf)
{
  __shared__ __align__(16) _Float16 ch[8][16][72];

  const int lane = threadIdx.x & 31;
  const int w    = threadIdx.x >> 5;
  const int task = blockIdx.x * 8 + w;
  const int t16  = task % NT16;
  const int h    = (task / NT16) % Hq;
  const int b    = task / (NT16 * Hq);
  const int l16  = lane & 15, grp = lane >> 4;

  const float* xrow = seq + (size_t)(b * Sq + t16 * 16 + l16) * Dq + h * DHq;
  v16h ax0, ax1;
#pragma unroll
  for (int o = 0; o < 8; ++o) {
    int d0 = grp * 8 + o;
    ax0[o]     = (_Float16)xrow[d0];
    ax0[8 + o] = (_Float16)xrow[16 + d0];
    ax1[o]     = (_Float16)xrow[32 + d0];
    ax1[8 + o] = (_Float16)xrow[48 + d0];
  }

  _Float16* cht = &ch[w][0][0];
  project_store<0>(lane, b, h, t16, ax0, ax1, wfrag, bq, qbuf, cht);
  project_store<1>(lane, b, h, t16, ax0, ax1, wfrag, bk, kbuf, cht);
  project_store<2>(lane, b, h, t16, ax0, ax1, wfrag, bv, vbuf, cht);
}

// ---------------------------------------------------------------------------
// Phase 2: flash attention.  One wave per 16-query tile; 128 keys/iteration:
// 16 WMMA for S = Q K^T, 16 WMMA for O += P V.  Row-sums are lane-partial
// (reduced once in the epilogue); running max uses a packed-f16 butterfly
// (16 bpermutes / 128 keys).
// ---------------------------------------------------------------------------
__global__ __launch_bounds__(256) void attn_kernel(
    const _Float16* __restrict__ qbuf, const _Float16* __restrict__ kbuf,
    const _Float16* __restrict__ vbuf, float* __restrict__ out)
{
  __shared__ __align__(16) _Float16 plds[8][16][136];

  const int lane = threadIdx.x & 31;
  const int w    = threadIdx.x >> 5;
  const int task = blockIdx.x * 8 + w;
  const int t16  = task % NT16;
  const int h    = (task / NT16) % Hq;
  const int b    = task / (NT16 * Hq);
  const int l16  = lane & 15, grp = lane >> 4;

  const _Float16* qb = qbuf + qidx(b, h, t16, 0, 0, 0);
  v16h qa0 = *(const v16h*)(qb + lane * 16);
  v16h qa1 = *(const v16h*)(qb + (32 + lane) * 16);

  v8f o0 = {}, o1 = {}, o2 = {}, o3 = {};
  float m[8], l[8];
#pragma unroll
  for (int i = 0; i < 8; ++i) { m[i] = -1e30f; l[i] = 0.f; }

  for (int j = 0; j < NT128; ++j) {
    // ---- scores for keys [128j, 128j+128): 8 C fragments, two halves ----
    const _Float16* kb = kbuf + qidx(b, h, 8 * j, 0, 0, 0);
    v8f c0 = {}, c1 = {}, c2 = {}, c3 = {}, c4 = {}, c5 = {}, c6 = {}, c7 = {};
    {
      v16h k0 = *(const v16h*)(kb + lane * 16);
      v16h k1 = *(const v16h*)(kb + 512 + lane * 16);
      v16h k2 = *(const v16h*)(kb + 1024 + lane * 16);
      v16h k3 = *(const v16h*)(kb + 1536 + lane * 16);
      c0 = wmma_f16(qa0, k0, c0);
      c1 = wmma_f16(qa0, k2, c1);
      c0 = wmma_f16(qa1, k1, c0);
      c1 = wmma_f16(qa1, k3, c1);
      v16h k4 = *(const v16h*)(kb + 2048 + lane * 16);
      v16h k5 = *(const v16h*)(kb + 2560 + lane * 16);
      v16h k6 = *(const v16h*)(kb + 3072 + lane * 16);
      v16h k7 = *(const v16h*)(kb + 3584 + lane * 16);
      c2 = wmma_f16(qa0, k4, c2);
      c3 = wmma_f16(qa0, k6, c3);
      c2 = wmma_f16(qa1, k5, c2);
      c3 = wmma_f16(qa1, k7, c3);
    }
    {
      v16h k0 = *(const v16h*)(kb + 4096 + lane * 16);
      v16h k1 = *(const v16h*)(kb + 4608 + lane * 16);
      v16h k2 = *(const v16h*)(kb + 5120 + lane * 16);
      v16h k3 = *(const v16h*)(kb + 5632 + lane * 16);
      c4 = wmma_f16(qa0, k0, c4);
      c5 = wmma_f16(qa0, k2, c5);
      c4 = wmma_f16(qa1, k1, c4);
      c5 = wmma_f16(qa1, k3, c5);
      v16h k4 = *(const v16h*)(kb + 6144 + lane * 16);
      v16h k5 = *(const v16h*)(kb + 6656 + lane * 16);
      v16h k6 = *(const v16h*)(kb + 7168 + lane * 16);
      v16h k7 = *(const v16h*)(kb + 7680 + lane * 16);
      c6 = wmma_f16(qa0, k4, c6);
      c7 = wmma_f16(qa0, k6, c7);
      c6 = wmma_f16(qa1, k5, c6);
      c7 = wmma_f16(qa1, k7, c7);
    }

    // ---- local row max over 128 keys ----
    float rmax[8];
#pragma unroll
    for (int i = 0; i < 8; ++i) {
      float a = fmaxf(fmaxf(c0[i], c1[i]), fmaxf(c2[i], c3[i]));
      float bmx = fmaxf(fmaxf(c4[i], c5[i]), fmaxf(c6[i], c7[i]));
      rmax[i] = fmaxf(a, bmx);
    }
    // ---- packed-f16 cross-lane max (4 steps x 4 packed regs) ----
    h2 pk[4];
#pragma unroll
    for (int p = 0; p < 4; ++p)
      pk[p] = __builtin_amdgcn_cvt_pkrtz(rmax[2 * p], rmax[2 * p + 1]);
#pragma unroll
    for (int p = 0; p < 4; ++p) {
      int vbits = __builtin_bit_cast(int, pk[p]);
#pragma unroll
      for (int msk = 1; msk <= 8; msk <<= 1) {
        int sbits = __shfl_xor(vbits, msk, 32);
        h2 a = __builtin_bit_cast(h2, vbits);
        h2 bb = __builtin_bit_cast(h2, sbits);
        vbits = __builtin_bit_cast(int, h2max(a, bb));
      }
      pk[p] = __builtin_bit_cast(h2, vbits);
    }
#pragma unroll
    for (int i = 0; i < 8; ++i) {
      float rm = (float)pk[i >> 1][i & 1];
      float mn    = fmaxf(m[i], rm);
      float alpha = __expf(m[i] - mn);
      m[i] = mn;
      float p0 = __expf(c0[i] - mn);
      float p1 = __expf(c1[i] - mn);
      float p2 = __expf(c2[i] - mn);
      float p3 = __expf(c3[i] - mn);
      float p4 = __expf(c4[i] - mn);
      float p5 = __expf(c5[i] - mn);
      float p6 = __expf(c6[i] - mn);
      float p7 = __expf(c7[i] - mn);
      l[i] = l[i] * alpha +
             (((p0 + p1) + (p2 + p3)) + ((p4 + p5) + (p6 + p7)));
      o0[i] *= alpha; o1[i] *= alpha; o2[i] *= alpha; o3[i] *= alpha;
      _Float16* pw = &plds[w][i + grp * 8][l16];
      pw[0]   = (_Float16)p0;
      pw[16]  = (_Float16)p1;
      pw[32]  = (_Float16)p2;
      pw[48]  = (_Float16)p3;
      pw[64]  = (_Float16)p4;
      pw[80]  = (_Float16)p5;
      pw[96]  = (_Float16)p6;
      pw[112] = (_Float16)p7;
    }
    asm volatile("s_wait_dscnt 0" ::: "memory");

    // ---- reload P as four A fragments (keys 0..31,...,96..127) ----
    const _Float16* pr = &plds[w][l16][0];
    const int hi8 = grp * 8;
    v16h pa0 = cat8(*(const v8h*)(pr + hi8),      *(const v8h*)(pr + 16 + hi8));
    v16h pa1 = cat8(*(const v8h*)(pr + 32 + hi8), *(const v8h*)(pr + 48 + hi8));
    v16h pa2 = cat8(*(const v8h*)(pr + 64 + hi8), *(const v8h*)(pr + 80 + hi8));
    v16h pa3 = cat8(*(const v8h*)(pr + 96 + hi8), *(const v8h*)(pr + 112 + hi8));

    // ---- O += P V over four 32-key V tiles ----
    const _Float16* vb = vbuf + vidx(b, h, 4 * j, 0, 0, 0);
    {
      v16h v0 = *(const v16h*)(vb + lane * 16);
      v16h v1 = *(const v16h*)(vb + 512 + lane * 16);
      v16h v2 = *(const v16h*)(vb + 1024 + lane * 16);
      v16h v3 = *(const v16h*)(vb + 1536 + lane * 16);
      o0 = wmma_f16(pa0, v0, o0);
      o1 = wmma_f16(pa0, v1, o1);
      o2 = wmma_f16(pa0, v2, o2);
      o3 = wmma_f16(pa0, v3, o3);
      v16h v4 = *(const v16h*)(vb + 2048 + lane * 16);
      v16h v5 = *(const v16h*)(vb + 2560 + lane * 16);
      v16h v6 = *(const v16h*)(vb + 3072 + lane * 16);
      v16h v7 = *(const v16h*)(vb + 3584 + lane * 16);
      o0 = wmma_f16(pa1, v4, o0);
      o1 = wmma_f16(pa1, v5, o1);
      o2 = wmma_f16(pa1, v6, o2);
      o3 = wmma_f16(pa1, v7, o3);
    }
    {
      v16h v0 = *(const v16h*)(vb + 4096 + lane * 16);
      v16h v1 = *(const v16h*)(vb + 4608 + lane * 16);
      v16h v2 = *(const v16h*)(vb + 5120 + lane * 16);
      v16h v3 = *(const v16h*)(vb + 5632 + lane * 16);
      o0 = wmma_f16(pa2, v0, o0);
      o1 = wmma_f16(pa2, v1, o1);
      o2 = wmma_f16(pa2, v2, o2);
      o3 = wmma_f16(pa2, v3, o3);
      v16h v4 = *(const v16h*)(vb + 6144 + lane * 16);
      v16h v5 = *(const v16h*)(vb + 6656 + lane * 16);
      v16h v6 = *(const v16h*)(vb + 7168 + lane * 16);
      v16h v7 = *(const v16h*)(vb + 7680 + lane * 16);
      o0 = wmma_f16(pa3, v4, o0);
      o1 = wmma_f16(pa3, v5, o1);
      o2 = wmma_f16(pa3, v6, o2);
      o3 = wmma_f16(pa3, v7, o3);
    }
  }

  // ---- epilogue: one cross-lane sum reduction, normalize, store fp32 ----
#pragma unroll
  for (int i = 0; i < 8; ++i) {
    l[i] += __shfl_xor(l[i], 1, 32);
    l[i] += __shfl_xor(l[i], 2, 32);
    l[i] += __shfl_xor(l[i], 4, 32);
    l[i] += __shfl_xor(l[i], 8, 32);
    float inv = 1.0f / l[i];
    int row = i + grp * 8;
    size_t base = (size_t)(b * Sq + t16 * 16 + row) * Dq + h * DHq;
    out[base + l16]      = o0[i] * inv;
    out[base + 16 + l16] = o1[i] * inv;
    out[base + 32 + l16] = o2[i] * inv;
    out[base + 48 + l16] = o3[i] * inv;
  }
}

extern "C" void kernel_launch(void* const* d_in, const int* in_sizes, int n_in,
                              void* d_out, int out_size, void* d_ws, size_t ws_size,
                              hipStream_t stream) {
  const float* seq = (const float*)d_in[0];
  const float* wq  = (const float*)d_in[1];
  const float* bq  = (const float*)d_in[2];
  const float* wk  = (const float*)d_in[3];
  const float* bk  = (const float*)d_in[4];
  const float* wv  = (const float*)d_in[5];
  const float* bv  = (const float*)d_in[6];
  float* out = (float*)d_out;

  _Float16* qbuf  = (_Float16*)d_ws;            // 12.6 MB
  _Float16* kbuf  = qbuf + QKV_ELEMS;           // 12.6 MB
  _Float16* vbuf  = kbuf + QKV_ELEMS;           // 12.6 MB
  _Float16* wfrag = vbuf + QKV_ELEMS;           // 0.3 MB

  wprep_kernel<<<36, 256, 0, stream>>>(wq, wk, wv, wfrag);
  qkv_kernel<<<(Bq * Hq * NT16) / 8, 256, 0, stream>>>(
      seq, wfrag, bq, bk, bv, qbuf, kbuf, vbuf);
  attn_kernel<<<(Bq * Hq * NT16) / 8, 256, 0, stream>>>(qbuf, kbuf, vbuf, out);
}
